// ResidualSparseBlock2d_26783416057884
// MI455X (gfx1250) — compile-verified, matrix operand order checked by reading, not run on hardware
//
#include <hip/hip_runtime.h>

typedef __attribute__((ext_vector_type(16))) _Float16 v16h;
typedef __attribute__((ext_vector_type(8)))  float    v8f;
typedef __attribute__((ext_vector_type(4)))  unsigned int u32x4;

union VFrag { v16h v; u32x4 u[2]; };

struct F4 { float x, y, z, w; };
struct F2 { float x, y; };

// ---------------------------------------------------------------------------
// Conversions: f32 -> f16 activations, f32 -> f16 transposed weights
// ---------------------------------------------------------------------------
__global__ void k_cvt_x(const float* __restrict__ x, _Float16* __restrict__ xh, long n) {
  long i = (long)blockIdx.x * blockDim.x + threadIdx.x;
  if (i < n) xh[i] = (_Float16)x[i];
}

// W1: [9][32][64] -> W1t [9][64][32]; W2: [9][64][64] -> W2t [9][64][64](T); Wd: [32][64] -> Wdt [64][32]
__global__ void k_cvt_w(const float* __restrict__ W1, const float* __restrict__ W2,
                        const float* __restrict__ Wd,
                        _Float16* __restrict__ W1t, _Float16* __restrict__ W2t,
                        _Float16* __restrict__ Wdt) {
  int i = blockIdx.x * blockDim.x + threadIdx.x;
  if (i < 9 * 64 * 32) {
    int k = i / 2048, r = i % 2048, co = r / 32, ci = r % 32;
    W1t[i] = (_Float16)W1[(k * 32 + ci) * 64 + co];
  }
  if (i < 9 * 64 * 64) {
    int k = i / 4096, r = i % 4096, co = r / 64, ci = r % 64;
    W2t[i] = (_Float16)W2[(k * 64 + ci) * 64 + co];
  }
  if (i < 64 * 32) {
    int co = i / 32, ci = i % 32;
    Wdt[i] = (_Float16)Wd[ci * 64 + co];
  }
}

__device__ __forceinline__ VFrag load_frag(const _Float16* p) {
  VFrag f;
  f.u[0] = *reinterpret_cast<const u32x4*>(p);
  f.u[1] = *reinterpret_cast<const u32x4*>(p + 16);
  return f;
}

// ---------------------------------------------------------------------------
// conv1: h1 = sum_k gather(x)@W1[k], id = x@Wd ; per-block BN partial sums
// block = 256 threads = 8 waves = 2 row-tiles(16 voxels) x 4 col-tiles(16 ch)
// Three independent WMMA accumulator chains (even-k, odd-k, 1x1 branch) so
// consecutive v_wmma ops have independent D registers (no serial hazard).
// ---------------------------------------------------------------------------
__global__ __launch_bounds__(256)
void k_conv1(const _Float16* __restrict__ xh, const int* __restrict__ nbr,
             const _Float16* __restrict__ W1t, const _Float16* __restrict__ Wdt,
             float* __restrict__ h1, float* __restrict__ idb,
             F4* __restrict__ part, int Nv) {
  const int lane = threadIdx.x & 31;
  const int wave = threadIdx.x >> 5;
  const int rt = wave >> 2;          // row tile 0..1
  const int ct = wave & 3;           // col tile 0..3
  const int half = lane >> 4;        // K-half select per ISA A/B layout
  const int l16 = lane & 15;
  const int baseh = half << 3;       // K offset in halfs (0 or 8)
  const int col = ct * 16 + l16;

  int vrow = blockIdx.x * 32 + rt * 16 + l16;
  if (vrow >= Nv) vrow = Nv - 1;

  v8f acc0 = {};   // even k chain
  v8f acc1 = {};   // odd k chain
  v8f accd = {};   // downsample chain

  // 1x1 branch (independent chain, interleaves with start of k-loop)
  {
    VFrag a = load_frag(xh + (long)vrow * 32 + baseh);
    VFrag b = load_frag(Wdt + col * 32 + baseh);
    accd = __builtin_amdgcn_wmma_f32_16x16x32_f16(false, a.v, false, b.v,
                                                  (short)0, accd, false, false);
  }

#pragma unroll
  for (int k = 0; k < 9; ++k) {
    const int idx = nbr[vrow * 9 + k];
    VFrag a = load_frag(xh + (long)idx * 32 + baseh);
    VFrag b = load_frag(W1t + (k * 64 + col) * 32 + baseh);
    if (k & 1)
      acc1 = __builtin_amdgcn_wmma_f32_16x16x32_f16(false, a.v, false, b.v,
                                                    (short)0, acc1, false, false);
    else
      acc0 = __builtin_amdgcn_wmma_f32_16x16x32_f16(false, a.v, false, b.v,
                                                    (short)0, acc0, false, false);
  }
  v8f acc = acc0 + acc1;

  // store C/D (lane l, reg r -> M = r + 8*half, N = l16) + local BN stats
  float s1 = 0.f, q1 = 0.f, sd = 0.f, qd = 0.f;
#pragma unroll
  for (int r = 0; r < 8; ++r) {
    const int m = blockIdx.x * 32 + rt * 16 + half * 8 + r;
    const float v1 = acc[r], vd = accd[r];
    if (m < Nv) {
      h1[(long)m * 64 + col] = v1;
      idb[(long)m * 64 + col] = vd;
    }
    s1 += v1; q1 += v1 * v1; sd += vd; qd += vd * vd;
  }
  // combine the two half-lanes (same channel, other 8 rows)
  s1 += __shfl_xor(s1, 16, 32);
  q1 += __shfl_xor(q1, 16, 32);
  sd += __shfl_xor(sd, 16, 32);
  qd += __shfl_xor(qd, 16, 32);

  __shared__ F4 sb[8][16];
  if (half == 0) { F4 t; t.x = s1; t.y = q1; t.z = sd; t.w = qd; sb[wave][l16] = t; }
  __syncthreads();
  if (threadIdx.x < 64) {
    const int ch = threadIdx.x;
    F4 p0 = sb[ch >> 4][ch & 15];
    F4 p1 = sb[(ch >> 4) + 4][ch & 15];
    F4 t; t.x = p0.x + p1.x; t.y = p0.y + p1.y; t.z = p0.z + p1.z; t.w = p0.w + p1.w;
    part[(long)blockIdx.x * 64 + ch] = t;
  }
}

// ---------------------------------------------------------------------------
// conv2: h2 = sum_k gather(out1)@W2[k] (Cin=64 -> 2 K-chunks per offset)
// One accumulator chain per Cin-chunk -> adjacent WMMAs are independent.
// ---------------------------------------------------------------------------
__global__ __launch_bounds__(256)
void k_conv2(const _Float16* __restrict__ yh, const int* __restrict__ nbr,
             const _Float16* __restrict__ W2t,
             float* __restrict__ h2, F2* __restrict__ part, int Nv) {
  const int lane = threadIdx.x & 31;
  const int wave = threadIdx.x >> 5;
  const int rt = wave >> 2;
  const int ct = wave & 3;
  const int half = lane >> 4;
  const int l16 = lane & 15;
  const int baseh = half << 3;
  const int col = ct * 16 + l16;

  int vrow = blockIdx.x * 32 + rt * 16 + l16;
  if (vrow >= Nv) vrow = Nv - 1;

  v8f acc0 = {};   // cc = 0 chain
  v8f acc1 = {};   // cc = 1 chain

#pragma unroll
  for (int k = 0; k < 9; ++k) {
    const int idx = nbr[vrow * 9 + k];
    const _Float16* row = yh + (long)idx * 64;
    const _Float16* wrow = W2t + (k * 64 + col) * 64;
    VFrag a0 = load_frag(row + baseh);
    VFrag b0 = load_frag(wrow + baseh);
    VFrag a1 = load_frag(row + 32 + baseh);
    VFrag b1 = load_frag(wrow + 32 + baseh);
    acc0 = __builtin_amdgcn_wmma_f32_16x16x32_f16(false, a0.v, false, b0.v,
                                                  (short)0, acc0, false, false);
    acc1 = __builtin_amdgcn_wmma_f32_16x16x32_f16(false, a1.v, false, b1.v,
                                                  (short)0, acc1, false, false);
  }
  v8f acc = acc0 + acc1;

  float s = 0.f, q = 0.f;
#pragma unroll
  for (int r = 0; r < 8; ++r) {
    const int m = blockIdx.x * 32 + rt * 16 + half * 8 + r;
    const float v = acc[r];
    if (m < Nv) h2[(long)m * 64 + col] = v;
    s += v; q += v * v;
  }
  s += __shfl_xor(s, 16, 32);
  q += __shfl_xor(q, 16, 32);

  __shared__ F2 sb[8][16];
  if (half == 0) { F2 t; t.x = s; t.y = q; sb[wave][l16] = t; }
  __syncthreads();
  if (threadIdx.x < 64) {
    const int ch = threadIdx.x;
    F2 p0 = sb[ch >> 4][ch & 15];
    F2 p1 = sb[(ch >> 4) + 4][ch & 15];
    F2 t; t.x = p0.x + p1.x; t.y = p0.y + p1.y;
    part[(long)blockIdx.x * 64 + ch] = t;
  }
}

// ---------------------------------------------------------------------------
// Deterministic reductions of per-block partials -> fused BN scale/bias
// coef: [0]=a1 [64]=b1 [128]=ad [192]=bd [256]=a2 [320]=b2 (per channel)
// ---------------------------------------------------------------------------
__global__ __launch_bounds__(256)
void k_red1(const F4* __restrict__ part, int nblk,
            const float* __restrict__ g1, const float* __restrict__ b1,
            const float* __restrict__ gd, const float* __restrict__ bd,
            float* __restrict__ coef, float invN) {
  __shared__ F4 red[256];
  const int ch = blockIdx.x;
  F4 s; s.x = s.y = s.z = s.w = 0.f;
  for (int i = threadIdx.x; i < nblk; i += 256) {
    F4 p = part[(long)i * 64 + ch];
    s.x += p.x; s.y += p.y; s.z += p.z; s.w += p.w;
  }
  red[threadIdx.x] = s;
  __syncthreads();
  for (int st = 128; st > 0; st >>= 1) {
    if (threadIdx.x < st) {
      F4 o = red[threadIdx.x + st];
      red[threadIdx.x].x += o.x; red[threadIdx.x].y += o.y;
      red[threadIdx.x].z += o.z; red[threadIdx.x].w += o.w;
    }
    __syncthreads();
  }
  if (threadIdx.x == 0) {
    F4 S = red[0];
    float m1 = S.x * invN, v1 = S.y * invN - m1 * m1;
    float a1 = g1[ch] * rsqrtf(v1 + 1e-5f);
    coef[ch] = a1; coef[64 + ch] = b1[ch] - a1 * m1;
    float md = S.z * invN, vd = S.w * invN - md * md;
    float ad = gd[ch] * rsqrtf(vd + 1e-5f);
    coef[128 + ch] = ad; coef[192 + ch] = bd[ch] - ad * md;
  }
}

__global__ __launch_bounds__(256)
void k_red2(const F2* __restrict__ part, int nblk,
            const float* __restrict__ g2, const float* __restrict__ b2,
            float* __restrict__ coef, float invN) {
  __shared__ F2 red[256];
  const int ch = blockIdx.x;
  F2 s; s.x = s.y = 0.f;
  for (int i = threadIdx.x; i < nblk; i += 256) {
    F2 p = part[(long)i * 64 + ch];
    s.x += p.x; s.y += p.y;
  }
  red[threadIdx.x] = s;
  __syncthreads();
  for (int st = 128; st > 0; st >>= 1) {
    if (threadIdx.x < st) {
      F2 o = red[threadIdx.x + st];
      red[threadIdx.x].x += o.x; red[threadIdx.x].y += o.y;
    }
    __syncthreads();
  }
  if (threadIdx.x == 0) {
    F2 S = red[0];
    float m2 = S.x * invN, v2 = S.y * invN - m2 * m2;
    float a2 = g2[ch] * rsqrtf(v2 + 1e-5f);
    coef[256 + ch] = a2; coef[320 + ch] = b2[ch] - a2 * m2;
  }
}

// ---------------------------------------------------------------------------
// Elementwise fusions
// ---------------------------------------------------------------------------
__global__ void k_bnrelu1(const float* __restrict__ h1, const float* __restrict__ coef,
                          _Float16* __restrict__ yh, long n) {
  long i = (long)blockIdx.x * blockDim.x + threadIdx.x;
  if (i >= n) return;
  int ch = (int)(i & 63);
  float v = coef[ch] * h1[i] + coef[64 + ch];
  yh[i] = (_Float16)fmaxf(v, 0.f);
}

__global__ void k_final(const float* __restrict__ h2, const float* __restrict__ idb,
                        const float* __restrict__ coef, float* __restrict__ out, long n) {
  long i = (long)blockIdx.x * blockDim.x + threadIdx.x;
  if (i >= n) return;
  int ch = (int)(i & 63);
  float v = coef[256 + ch] * h2[i] + coef[320 + ch]   // BN2(h2)
          + coef[128 + ch] * idb[i] + coef[192 + ch]; // BN(identity)
  out[i] = fmaxf(v, 0.f);
}

// ---------------------------------------------------------------------------
extern "C" void kernel_launch(void* const* d_in, const int* in_sizes, int n_in,
                              void* d_out, int out_size, void* d_ws, size_t ws_size,
                              hipStream_t stream) {
  const float* x  = (const float*)d_in[0];
  const float* W1 = (const float*)d_in[1];
  const float* g1 = (const float*)d_in[2];
  const float* b1 = (const float*)d_in[3];
  const float* W2 = (const float*)d_in[4];
  const float* g2 = (const float*)d_in[5];
  const float* b2 = (const float*)d_in[6];
  const float* Wd = (const float*)d_in[7];
  const float* gd = (const float*)d_in[8];
  const float* bd = (const float*)d_in[9];
  const int*  nbr = (const int*)d_in[10];

  const int Nv = in_sizes[0] / 32;
  const int nblk = (Nv + 31) / 32;

  char* p = (char*)d_ws;
  auto take = [&](size_t bytes) { char* r = p; p += (bytes + 255) & ~(size_t)255; return r; };
  _Float16* xh  = (_Float16*)take((size_t)Nv * 32 * 2);
  _Float16* W1t = (_Float16*)take((size_t)9 * 64 * 32 * 2);
  _Float16* W2t = (_Float16*)take((size_t)9 * 64 * 64 * 2);
  _Float16* Wdt = (_Float16*)take((size_t)64 * 32 * 2);
  float*    h12 = (float*)take((size_t)Nv * 64 * 4);   // h1, later reused as h2
  float*    idb = (float*)take((size_t)Nv * 64 * 4);   // identity pre-BN
  _Float16* yh  = (_Float16*)take((size_t)Nv * 64 * 2);
  F4*     part1 = (F4*)take((size_t)nblk * 64 * sizeof(F4));
  F2*     part2 = (F2*)take((size_t)nblk * 64 * sizeof(F2));
  float*   coef = (float*)take(384 * 4);
  (void)ws_size; (void)n_in; (void)out_size;

  const float invN = 1.0f / (float)Nv;
  const long nx = (long)Nv * 32;
  const long ny = (long)Nv * 64;

  k_cvt_x<<<(int)((nx + 255) / 256), 256, 0, stream>>>(x, xh, nx);
  k_cvt_w<<<(9 * 64 * 64 + 255) / 256, 256, 0, stream>>>(W1, W2, Wd, W1t, W2t, Wdt);
  k_conv1<<<nblk, 256, 0, stream>>>(xh, nbr, W1t, Wdt, h12, idb, part1, Nv);
  k_red1<<<64, 256, 0, stream>>>(part1, nblk, g1, b1, gd, bd, coef, invN);
  k_bnrelu1<<<(int)((ny + 255) / 256), 256, 0, stream>>>(h12, coef, yh, ny);
  k_conv2<<<nblk, 256, 0, stream>>>(yh, nbr, W2t, h12, part2, Nv);
  k_red2<<<64, 256, 0, stream>>>(part2, nblk, g2, b2, coef, invN);
  k_final<<<(int)((ny + 255) / 256), 256, 0, stream>>>(h12, idb, coef, (float*)d_out, ny);
}
